// MultiheadAttentionFFN_54417235640330
// MI455X (gfx1250) — matrix operand order repaired and checked by reading, not compiled, not run on hardware
//
#include <hip/hip_runtime.h>
#include <hip/hip_bf16.h>
#include <math.h>

// ---------------------------------------------------------------------------
// Types for CDNA5 WMMA (gfx1250, wave32)
// ---------------------------------------------------------------------------
typedef __bf16 bf16_t;
typedef __attribute__((ext_vector_type(16))) __bf16 v16bf;
typedef __attribute__((ext_vector_type(8)))  __bf16 v8bf;
typedef __attribute__((ext_vector_type(8)))  float  v8f;
typedef __attribute__((ext_vector_type(4)))  unsigned int u32x4;
typedef __attribute__((ext_vector_type(8)))  unsigned int u32x8;

#define T_DIM 2048
#define B_DIM 2
#define E_DIM 512
#define H_DIM 8
#define D_DIM 64
#define LDTB  (B_DIM * E_DIM)   // 1024

// ---------------------------------------------------------------------------
// TDM helpers (CDNA5 ISA/§8): 2D tile load to LDS.
// Descriptor built per ISA bitfield tables; issued from one wave per block.
// ---------------------------------------------------------------------------
__device__ __forceinline__ unsigned lds_off32(const void* p) {
  // Generic LDS address encodes the wave-relative LDS byte offset in [31:0].
  return (unsigned)(unsigned long long)p;
}

__device__ __forceinline__ void tdm_load_2d(const bf16_t* gsrc, unsigned lds_byte,
                                            unsigned width_elems /* dim0 len+stride */,
                                            unsigned tile_w, unsigned tile_h) {
  unsigned long long ga = (unsigned long long)gsrc;
  u32x4 g0;
  g0[0] = 1u;                                            // count=1, user desc
  g0[1] = lds_byte;                                      // lds_addr
  g0[2] = (unsigned)ga;                                  // global_addr[31:0]
  g0[3] = (unsigned)((ga >> 32) & 0x01FFFFFFu) | (2u << 30); // addr[56:32] | type=2
  const unsigned td0 = width_elems;                      // tensor_dim0
  const unsigned td1 = 0x7FFFFFFFu;                      // tall enough (tiles in-bounds)
  const unsigned st0 = width_elems;                      // tensor_dim0_stride
  u32x8 g1;
  g1[0] = (1u << 16);                                    // data_size=1 -> 2 bytes
  g1[1] = (td0 & 0xFFFFu) << 16;                         // tensor_dim0[15:0]
  g1[2] = (td0 >> 16) | ((td1 & 0xFFFFu) << 16);         // dim0 hi | dim1 lo
  g1[3] = (td1 >> 16) | (tile_w << 16);                  // dim1 hi | tile_dim0
  g1[4] = tile_h;                                        // tile_dim1 (tile_dim2=0)
  g1[5] = st0;                                           // dim0_stride[31:0]
  g1[6] = 0u;                                            // stride hi | dim1_stride lo
  g1[7] = 0u;
  asm volatile("tensor_load_to_lds %0, %1" :: "s"(g0), "s"(g1) : "memory");
}

__device__ __forceinline__ void tdm_wait() {
#if __has_builtin(__builtin_amdgcn_s_wait_tensorcnt)
  __builtin_amdgcn_s_wait_tensorcnt(0);
#else
  asm volatile("s_wait_tensorcnt 0x0" ::: "memory");
#endif
}

// ---------------------------------------------------------------------------
// f32 -> bf16 conversion
// ---------------------------------------------------------------------------
__global__ void k_cvt_bf16(const float* __restrict__ in, bf16_t* __restrict__ out, int n) {
  int i = blockIdx.x * blockDim.x + threadIdx.x;
  if (i < n) out[i] = (bf16_t)in[i];
}

// ---------------------------------------------------------------------------
// bf16 WMMA GEMM with TDM-staged B tiles.
// Block: 256 thr = 8 waves -> 128(M) x 64(N) tile. K % 32 == 0, M % 128 == 0.
// B tile (32x64 bf16, 4 KB) double-buffered in LDS via tensor_load_to_lds,
// prefetched one k-step ahead; A fragments register-pipelined.
// ---------------------------------------------------------------------------
__global__ void k_gemm_bf16(const bf16_t* __restrict__ A, int lda,
                            const bf16_t* __restrict__ B, int ldb,
                            const float* __restrict__ bias,
                            const float* __restrict__ res, int ldres,
                            void* __restrict__ outp, int ldout,
                            int M, int N, int K, int relu, int out_bf16) {
  __shared__ __align__(128) bf16_t bstage[2][32][64];
  const int lane = threadIdx.x & 31;
  const int wid  = threadIdx.x >> 5;
  const int tiles_n = N >> 6;
  const int m0 = (blockIdx.x / tiles_n) * 128 + wid * 16;
  const int n0 = (blockIdx.x % tiles_n) * 64;

  v8f acc[4];
#pragma unroll
  for (int t = 0; t < 4; ++t) { v8f z = {}; acc[t] = z; }

  const int mrow = m0 + (lane & 15);
  const int kb   = (lane >> 4) << 3;
  const bf16_t* Arow = A + (unsigned)(mrow * lda);

  // stage first B tile
  if (wid == 0) {
    tdm_load_2d(B + n0, lds_off32(&bstage[0][0][0]), (unsigned)ldb, 64u, 32u);
    tdm_wait();
  }
  __syncthreads();

  // preload first A fragment
  v8bf alo = *(const v8bf*)(Arow + kb);
  v8bf ahi = *(const v8bf*)(Arow + 16 + kb);

  int cur = 0;
  for (int k0 = 0; k0 < K; k0 += 32) {
    // prefetch next B tile via TDM (overlaps with this step's WMMAs)
    if (wid == 0 && k0 + 32 < K)
      tdm_load_2d(B + (unsigned)((k0 + 32) * ldb) + n0,
                  lds_off32(&bstage[cur ^ 1][0][0]), (unsigned)ldb, 64u, 32u);

    // prefetch next A fragment
    const int kn = (k0 + 32 < K) ? k0 + 32 : k0;
    v8bf alo2 = *(const v8bf*)(Arow + kn + kb);
    v8bf ahi2 = *(const v8bf*)(Arow + kn + 16 + kb);

    v16bf af;
#pragma unroll
    for (int i = 0; i < 8; ++i) { af[i] = alo[i]; af[8 + i] = ahi[i]; }

    // all four B fragments from LDS (lane = wmma-N col; elements = K rows)
    v16bf bfr[4];
#pragma unroll
    for (int t = 0; t < 4; ++t) {
      const bf16_t* bc = &bstage[cur][0][t * 16 + (lane & 15)];
#pragma unroll
      for (int i = 0; i < 8; ++i) { bfr[t][i] = bc[(kb + i) * 64]; bfr[t][8 + i] = bc[(16 + kb + i) * 64]; }
    }
#pragma unroll
    for (int t = 0; t < 4; ++t)
      acc[t] = __builtin_amdgcn_wmma_f32_16x16x32_bf16(false, af, false, bfr[t],
                                                       (short)0, acc[t], false, false);
    alo = alo2; ahi = ahi2;
    if (wid == 0) tdm_wait();
    __syncthreads();
    cur ^= 1;
  }

  const int r0 = (lane >> 4) << 3;
  const int nc = lane & 15;
#pragma unroll
  for (int t = 0; t < 4; ++t) {
    const int n = n0 + t * 16 + nc;
    const float bv = bias ? bias[n] : 0.0f;
#pragma unroll
    for (int j = 0; j < 8; ++j) {
      const int row = m0 + r0 + j;
      float val = acc[t][j] + bv;
      if (relu) val = val > 0.0f ? val : 0.0f;
      if (res)  val += res[(size_t)row * ldres + n];
      if (out_bf16) ((bf16_t*)outp)[(size_t)row * ldout + n] = (bf16_t)val;
      else          ((float*)outp)[(size_t)row * ldout + n] = val;
    }
  }
}

// ---------------------------------------------------------------------------
// Row LayerNorm -> bf16. kt_mode=1 writes K transposed [B*H, D, T] for the
// attention kernel (contiguous K^T WMMA fragments).
// ---------------------------------------------------------------------------
__global__ void k_ln_rows(const float* __restrict__ in, const float* __restrict__ g,
                          const float* __restrict__ bp, bf16_t* __restrict__ out,
                          int rows, int L, float alpha, int kt_mode) {
  const int lane = threadIdx.x & 31;
  const int row  = blockIdx.x * (blockDim.x >> 5) + (threadIdx.x >> 5);
  if (row >= rows) return;
  const float* x = in + (size_t)row * L;
  float s = 0.0f, s2 = 0.0f;
  for (int i = lane; i < L; i += 32) { float v = x[i]; s += v; s2 += v * v; }
#pragma unroll
  for (int m = 16; m >= 1; m >>= 1) { s += __shfl_xor(s, m, 32); s2 += __shfl_xor(s2, m, 32); }
  const float mu  = s / (float)L;
  const float var = s2 / (float)L - mu * mu;
  const float rs  = rsqrtf(var + 1e-5f);
  if (kt_mode) {
    // row = t*B + b (L==E); element i = h*D + d -> kT[((b*H+h)*D + d)*T + t]
    const int t = row >> 1, b = row & 1;
    for (int i = lane; i < L; i += 32) {
      float v = (x[i] - mu) * rs * g[i] + bp[i];
      const int h = i >> 6, d = i & 63;
      out[(size_t)(((b * H_DIM + h) * D_DIM + d)) * T_DIM + t] = (bf16_t)(v * alpha);
    }
  } else {
    bf16_t* y = out + (size_t)row * L;
    for (int i = lane; i < L; i += 32) {
      float v = (x[i] - mu) * rs * g[i] + bp[i];
      y[i] = (bf16_t)(v * alpha);
    }
  }
}

// ---------------------------------------------------------------------------
// Elementwise gate multiply -> bf16
// ---------------------------------------------------------------------------
__global__ void k_mul_bf16(const float* __restrict__ a, const float* __restrict__ b,
                           bf16_t* __restrict__ out, int n) {
  int i = blockIdx.x * blockDim.x + threadIdx.x;
  if (i < n) out[i] = (bf16_t)(a[i] * b[i]);
}

// ---------------------------------------------------------------------------
// Flash attention. q,v: [T,B,E] bf16 (q pre-scaled by D^-0.5);
// kT: [B*H, D, T] bf16 (transposed). o: [T,B,H,D] f32.
// One wave per 16-row t tile; stream keys/values in 32-column chunks.
// ---------------------------------------------------------------------------
#define AW 4
__global__ void k_flash_attn(const bf16_t* __restrict__ q, const bf16_t* __restrict__ kT,
                             const bf16_t* __restrict__ vmat, float* __restrict__ o) {
  __shared__ __align__(16) bf16_t plds[AW][16][32];
  const int lane = threadIdx.x & 31;
  const int wid  = threadIdx.x >> 5;
  const int bh   = blockIdx.y;            // b*H + h
  const int base = bh * D_DIM;            // b*E + h*D
  const int t0   = (blockIdx.x * AW + wid) * 16;

  const int mrow = t0 + (lane & 15);
  const int kb   = (lane >> 4) << 3;

  v16bf qf[2];
#pragma unroll
  for (int h = 0; h < 2; ++h) {
    const bf16_t* p = q + (size_t)mrow * LDTB + base + h * 32;
    v8bf lo = *(const v8bf*)(p + kb);
    v8bf hi = *(const v8bf*)(p + 16 + kb);
#pragma unroll
    for (int i = 0; i < 8; ++i) { qf[h][i] = lo[i]; qf[h][8 + i] = hi[i]; }
  }

  float mr[8], lr[8], sc[8];
  v8f oacc[4];
#pragma unroll
  for (int j = 0; j < 8; ++j) { mr[j] = -1e30f; lr[j] = 0.0f; }
#pragma unroll
  for (int t = 0; t < 4; ++t) { v8f z = {}; oacc[t] = z; }

  const bf16_t* kbase = kT + (size_t)bh * D_DIM * T_DIM;

  for (int s0 = 0; s0 < T_DIM; s0 += 32) {
    v8f sacc[2];
#pragma unroll
    for (int h = 0; h < 2; ++h) { v8f z = {}; sacc[h] = z; }
#pragma unroll
    for (int half = 0; half < 2; ++half) {
#pragma unroll
      for (int dk = 0; dk < 2; ++dk) {
        // K^T fragment: lane = wmma-N col (key s); elements = head dim d (strided)
        const bf16_t* kr = kbase + (size_t)(dk * 32) * T_DIM + s0 + half * 16 + (lane & 15);
        v16bf bfr;
#pragma unroll
        for (int i = 0; i < 8; ++i) { bfr[i] = kr[(size_t)(kb + i) * T_DIM]; bfr[8 + i] = kr[(size_t)(16 + kb + i) * T_DIM]; }
        sacc[half] = __builtin_amdgcn_wmma_f32_16x16x32_bf16(false, qf[dk], false, bfr,
                                                             (short)0, sacc[half], false, false);
      }
    }

    // online softmax (row stats match C-fragment row layout)
#pragma unroll
    for (int j = 0; j < 8; ++j) {
      float vmax = fmaxf(sacc[0][j], sacc[1][j]);
#pragma unroll
      for (int m = 1; m < 16; m <<= 1) vmax = fmaxf(vmax, __shfl_xor(vmax, m, 32));
      const float nm = fmaxf(mr[j], vmax);
      sc[j] = __expf(mr[j] - nm);
      mr[j] = nm;
      const float p0 = __expf(sacc[0][j] - nm);
      const float p1 = __expf(sacc[1][j] - nm);
      float rsum = p0 + p1;
#pragma unroll
      for (int m = 1; m < 16; m <<= 1) rsum += __shfl_xor(rsum, m, 32);
      lr[j] = lr[j] * sc[j] + rsum;
      sacc[0][j] = p0; sacc[1][j] = p1;
    }

    // P tile -> LDS (bf16) -> A fragment
    __syncthreads();
    {
      const int r0 = (lane >> 4) << 3;
      const int c  = lane & 15;
#pragma unroll
      for (int j = 0; j < 8; ++j) {
        plds[wid][r0 + j][c]      = (bf16_t)sacc[0][j];
        plds[wid][r0 + j][16 + c] = (bf16_t)sacc[1][j];
      }
    }
    __syncthreads();
    v16bf pf;
    {
      const bf16_t* pr = &plds[wid][lane & 15][0];
      v8bf lo = *(const v8bf*)(pr + kb);
      v8bf hi = *(const v8bf*)(pr + 16 + kb);
#pragma unroll
      for (int i = 0; i < 8; ++i) { pf[i] = lo[i]; pf[8 + i] = hi[i]; }
    }

    // O = diag(sc)*O + P @ V
#pragma unroll
    for (int t = 0; t < 4; ++t) {
      const bf16_t* vp = vmat + (size_t)s0 * LDTB + base + t * 16 + (lane & 15);
      v16bf vf;
#pragma unroll
      for (int i = 0; i < 8; ++i) { vf[i] = vp[(size_t)(kb + i) * LDTB]; vf[8 + i] = vp[(size_t)(16 + kb + i) * LDTB]; }
      v8f c = oacc[t];
#pragma unroll
      for (int j = 0; j < 8; ++j) c[j] *= sc[j];
      oacc[t] = __builtin_amdgcn_wmma_f32_16x16x32_bf16(false, pf, false, vf,
                                                        (short)0, c, false, false);
    }
  }

  {
    const int r0 = (lane >> 4) << 3;
    const int c  = lane & 15;
#pragma unroll
    for (int t = 0; t < 4; ++t)
#pragma unroll
      for (int j = 0; j < 8; ++j) {
        const int row = t0 + r0 + j;
        o[(size_t)row * LDTB + base + t * 16 + c] = oacc[t][j] / lr[j];
      }
  }
}

// ---------------------------------------------------------------------------
// Host-side orchestration
// ---------------------------------------------------------------------------
static inline size_t al256(size_t x) { return (x + 255) & ~(size_t)255; }

extern "C" void kernel_launch(void* const* d_in, const int* in_sizes, int n_in,
                              void* d_out, int out_size, void* d_ws, size_t ws_size,
                              hipStream_t stream) {
  const float* x   = (const float*)d_in[0];
  const float* Wq  = (const float*)d_in[1];  const float* bq = (const float*)d_in[2];
  const float* Wk  = (const float*)d_in[3];  const float* bk = (const float*)d_in[4];
  const float* Wv  = (const float*)d_in[5];  const float* bv = (const float*)d_in[6];
  const float* Wu  = (const float*)d_in[7];  const float* bu = (const float*)d_in[8];
  const float* Wo  = (const float*)d_in[9];  const float* bo = (const float*)d_in[10];
  const float* W1  = (const float*)d_in[11]; const float* b1 = (const float*)d_in[12];
  const float* W2  = (const float*)d_in[13]; const float* b2 = (const float*)d_in[14];
  const float* q_g = (const float*)d_in[15]; const float* q_b = (const float*)d_in[16];
  const float* k_g = (const float*)d_in[17]; const float* k_b = (const float*)d_in[18];
  const float* f_g = (const float*)d_in[19]; const float* f_b = (const float*)d_in[20];

  const int M    = T_DIM * B_DIM;           // 4096
  const int NTOK = M * E_DIM;               // 2,097,152
  const int MF   = T_DIM * B_DIM * H_DIM;   // 32768 FFN rows

  size_t off = 0;
  auto take = [&](size_t bytes) { size_t o = off; off = al256(off + bytes); return o; };
  char* ws = (char*)d_ws;
  bf16_t* xb   = (bf16_t*)(ws + take((size_t)NTOK * 2));
  bf16_t* wqb  = (bf16_t*)(ws + take((size_t)E_DIM * E_DIM * 2));
  bf16_t* wkb  = (bf16_t*)(ws + take((size_t)E_DIM * E_DIM * 2));
  bf16_t* wvb  = (bf16_t*)(ws + take((size_t)E_DIM * E_DIM * 2));
  bf16_t* wub  = (bf16_t*)(ws + take((size_t)E_DIM * E_DIM * 2));
  bf16_t* wob  = (bf16_t*)(ws + take((size_t)E_DIM * E_DIM * 2));
  bf16_t* w1b  = (bf16_t*)(ws + take((size_t)D_DIM * 4 * D_DIM * 2));
  bf16_t* w2b  = (bf16_t*)(ws + take((size_t)4 * D_DIM * D_DIM * 2));
  float*  pre  = (float*) (ws + take((size_t)NTOK * 4));   // reused q_pre -> k_pre
  bf16_t* qb2  = (bf16_t*)(ws + take((size_t)NTOK * 2));
  bf16_t* kTt  = (bf16_t*)(ws + take((size_t)NTOK * 2));   // K transposed [B*H, D, T]
  bf16_t* vb2  = (bf16_t*)(ws + take((size_t)NTOK * 2));
  float*  gate = (float*) (ws + take((size_t)NTOK * 4));
  float*  obuf = (float*) (ws + take((size_t)NTOK * 4));
  bf16_t* lno  = (bf16_t*)(ws + take((size_t)NTOK * 2));
  bf16_t* h1   = (bf16_t*)(ws + take((size_t)MF * 4 * D_DIM * 2));
  float*  offn = (float*) (ws + take((size_t)NTOK * 4));
  bf16_t* yb   = (bf16_t*)(ws + take((size_t)NTOK * 2));
  (void)ws_size; (void)n_in; (void)in_sizes; (void)out_size;

  // 1) f32 -> bf16 conversions
  k_cvt_bf16<<<(NTOK + 255) / 256, 256, 0, stream>>>(x, xb, NTOK);
  const int NW = E_DIM * E_DIM;
  k_cvt_bf16<<<(NW + 255) / 256, 256, 0, stream>>>(Wq, wqb, NW);
  k_cvt_bf16<<<(NW + 255) / 256, 256, 0, stream>>>(Wk, wkb, NW);
  k_cvt_bf16<<<(NW + 255) / 256, 256, 0, stream>>>(Wv, wvb, NW);
  k_cvt_bf16<<<(NW + 255) / 256, 256, 0, stream>>>(Wu, wub, NW);
  k_cvt_bf16<<<(NW + 255) / 256, 256, 0, stream>>>(Wo, wob, NW);
  const int NW1 = D_DIM * 4 * D_DIM;
  k_cvt_bf16<<<(NW1 + 255) / 256, 256, 0, stream>>>(W1, w1b, NW1);
  k_cvt_bf16<<<(NW1 + 255) / 256, 256, 0, stream>>>(W2, w2b, NW1);

  auto gemm = [&](const bf16_t* A, int lda, const bf16_t* B, int ldb,
                  const float* bias, const float* res, int ldres,
                  void* out, int ldout, int m, int n, int k, int relu, int obf) {
    const int tiles = (m >> 7) * (n >> 6);   // 128x64 block tiles
    k_gemm_bf16<<<tiles, 256, 0, stream>>>(A, lda, B, ldb, bias, res, ldres,
                                           out, ldout, m, n, k, relu, obf);
  };

  // 2) q = LN(x@Wq+bq)*D^-0.5 ; k = LN(x@Wk+bk) (transposed) ; v ; gate
  gemm(xb, E_DIM, wqb, E_DIM, bq, nullptr, 0, pre, E_DIM, M, E_DIM, E_DIM, 0, 0);
  k_ln_rows<<<(M + 7) / 8, 256, 0, stream>>>(pre, q_g, q_b, qb2, M, E_DIM, 0.125f, 0);
  gemm(xb, E_DIM, wkb, E_DIM, bk, nullptr, 0, pre, E_DIM, M, E_DIM, E_DIM, 0, 0);
  k_ln_rows<<<(M + 7) / 8, 256, 0, stream>>>(pre, k_g, k_b, kTt, M, E_DIM, 1.0f, 1);
  gemm(xb, E_DIM, wvb, E_DIM, bv, nullptr, 0, vb2, E_DIM, M, E_DIM, E_DIM, 0, 1);
  gemm(xb, E_DIM, wub, E_DIM, bu, nullptr, 0, gate, E_DIM, M, E_DIM, E_DIM, 0, 0);

  // 3) attention -> obuf [T,B,H,D]
  dim3 agrid(T_DIM / 16 / AW, B_DIM * H_DIM);
  k_flash_attn<<<agrid, 32 * AW, 0, stream>>>(qb2, kTt, vb2, obuf);

  // 4) per-head FFN with residual
  k_ln_rows<<<(MF + 7) / 8, 256, 0, stream>>>(obuf, f_g, f_b, lno, MF, D_DIM, 1.0f, 0);
  gemm(lno, D_DIM, w1b, 4 * D_DIM, b1, nullptr, 0, h1, 4 * D_DIM,
       MF, 4 * D_DIM, D_DIM, 1, 1);
  gemm(h1, 4 * D_DIM, w2b, D_DIM, b2, obuf, D_DIM, offn, D_DIM,
       MF, D_DIM, 4 * D_DIM, 0, 0);

  // 5) gate multiply and output projection
  k_mul_bf16<<<(NTOK + 255) / 256, 256, 0, stream>>>(offn, gate, yb, NTOK);
  gemm(yb, E_DIM, wob, E_DIM, bo, nullptr, 0, (float*)d_out, E_DIM,
       M, E_DIM, E_DIM, 0, 0);
}
